// GlobalConstraintDecomposer_24455543783627
// MI455X (gfx1250) — matrix operand order repaired and checked by reading, not compiled, not run on hardware
//
#include <hip/hip_runtime.h>
#include <hip/hip_bf16.h>

// ---------------------------------------------------------------------------
// Seq2seq stacked-LSTM forward on gfx1250 (MI455X), bf16 WMMA everywhere.
//   B=256, T_IN=256, D=512, H=1024, V=512, T_OUT=64, L=2, 4H=4096
// GEMMs use v_wmma_f32_16x16x32_bf16 with double-buffered LDS filled by
// global_load_async_to_lds_b128 (ASYNCcnt), overlapping DMA with WMMA issue.
// ---------------------------------------------------------------------------

typedef __attribute__((ext_vector_type(16))) __bf16 v16bf;
typedef __attribute__((ext_vector_type(8)))  float  v8f;

// ---- helpers --------------------------------------------------------------

__device__ __forceinline__ __bf16 f2bf(float f) {
  union { float f; unsigned u; } v; v.f = f;
  unsigned r = v.u + 0x7FFFu + ((v.u >> 16) & 1u);   // round-to-nearest-even
  unsigned short h = (unsigned short)(r >> 16);
  return __builtin_bit_cast(__bf16, h);
}
__device__ __forceinline__ float bf2f(__bf16 b) {
  unsigned short h = __builtin_bit_cast(unsigned short, b);
  unsigned u = ((unsigned)h) << 16;
  return __builtin_bit_cast(float, u);
}
__device__ __forceinline__ float sigm(float x) {
  return 1.0f / (1.0f + __expf(-x));
}
__device__ __forceinline__ float tanh_fast(float x) {
  x = fminf(15.0f, fmaxf(-15.0f, x));
  float e = __expf(2.0f * x);
  return (e - 1.0f) / (e + 1.0f);
}

// Raw 32-bit LDS address from a generic shared pointer (ISA: LDS addr = addr[31:0]).
__device__ __forceinline__ unsigned lds_lo(const void* p) {
  return (unsigned)(unsigned long long)p;
}
// Async DMA: 16 bytes global -> LDS, per lane. Tracked by ASYNCcnt.
__device__ __forceinline__ void async_ld_b128(unsigned lds, const void* g) {
  asm volatile("global_load_async_to_lds_b128 %0, %1, off"
               :: "v"(lds), "v"(g) : "memory");
}
__device__ __forceinline__ void wait_async0() {
  asm volatile("s_wait_asynccnt 0x0" ::: "memory");
}

// ---- fp32 -> bf16 strided convert/pack (also used for weight packing) -----

__global__ __launch_bounds__(256)
void cvt_f32_bf16(__bf16* __restrict__ dst, int ldd, int doff,
                  const float* __restrict__ src, int lds_, int soff,
                  int rows, int cols) {
  int i = blockIdx.x * blockDim.x + threadIdx.x;
  if (i >= rows * cols) return;
  int r = i / cols, c = i % cols;
  dst[(size_t)r * ldd + doff + c] = f2bf(src[(size_t)r * lds_ + soff + c]);
}

__global__ __launch_bounds__(256)
void zero_bf16_strided(__bf16* __restrict__ dst, int ldd, int doff,
                       int rows, int cols) {
  int i = blockIdx.x * blockDim.x + threadIdx.x;
  if (i >= rows * cols) return;
  int r = i / cols, c = i % cols;
  dst[(size_t)r * ldd + doff + c] = __builtin_bit_cast(__bf16, (unsigned short)0);
}

__global__ __launch_bounds__(256)
void fill_f32(float* __restrict__ p, float v, int n) {
  int i = blockIdx.x * blockDim.x + threadIdx.x;
  if (i < n) p[i] = v;
}

// ---- gate-buffer initializers ---------------------------------------------

__global__ __launch_bounds__(256)
void gates_init_bias(float* __restrict__ C, int ldc,
                     const float* __restrict__ bias, int Bn, int Nn) {
  int i = blockIdx.x * blockDim.x + threadIdx.x;
  if (i >= Bn * Nn) return;
  int b = i / Nn, n = i % Nn;
  C[(size_t)b * ldc + n] = bias[n];
}

// Decoder layer-0 input GEMM collapses to a column gather (one-hot input):
//   gates[b,n] = bias[n] + Wih[n, idx[b]]      (or just bias at t==0)
__global__ __launch_bounds__(256)
void gates_init_gather(float* __restrict__ gates, int ldc,
                       const float* __restrict__ bias,
                       const __bf16* __restrict__ Wih, int Vn,
                       const int* __restrict__ idx, int useIdx,
                       int Bn, int Nn) {
  int i = blockIdx.x * blockDim.x + threadIdx.x;
  if (i >= Bn * Nn) return;
  int b = i / Nn, n = i % Nn;
  float v = bias[n];
  if (useIdx) v += bf2f(Wih[(size_t)n * Vn + idx[b]]);
  gates[(size_t)b * ldc + n] = v;
}

// ---- WMMA GEMM: C[M,N] += A[M,K] @ W[N,K]^T  (bf16 in, fp32 accumulate) ----
// block = 256 threads = 8 wave32; block tile 128x128; wave tile 32x64;
// K stepped by 32 -> v_wmma_f32_16x16x32_bf16. Double-buffered LDS filled by
// async global->LDS DMA so chunk i+1 streams in while chunk i multiplies.
// Requires M%128==0, N%128==0, K%32==0, lda/ldw multiples of 8 - all true here.

__global__ __launch_bounds__(256)
void gemm_bf16_wmma(const __bf16* __restrict__ A, int lda,
                    const __bf16* __restrict__ W, int ldw,
                    float* __restrict__ C, int ldc,
                    int M, int N, int K) {
  __shared__ __bf16 As[2][128][40];   // +8 pad: avoids LDS bank conflicts
  __shared__ __bf16 Ws[2][128][40];

  const int tid  = threadIdx.x;
  const int lane = tid & 31;
  const int wv   = tid >> 5;
  const int wm   = (wv & 3) << 5;   // wave M offset: 0/32/64/96
  const int wn   = (wv >> 2) << 6;  // wave N offset: 0/64
  const int bm   = blockIdx.y * 128;
  const int bn   = blockIdx.x * 128;
  const int lr   = tid >> 1;        // staging row 0..127
  const int lc   = (tid & 1) << 4;  // staging col 0 or 16
  const int g    = lane >> 4;       // half-wave group
  const int l16  = lane & 15;

  const __bf16* gA = A + (size_t)(bm + lr) * lda + lc;
  const __bf16* gW = W + (size_t)(bn + lr) * ldw + lc;
  unsigned lA[2], lW[2];
  lA[0] = lds_lo(&As[0][lr][lc]);  lA[1] = lds_lo(&As[1][lr][lc]);
  lW[0] = lds_lo(&Ws[0][lr][lc]);  lW[1] = lds_lo(&Ws[1][lr][lc]);

  // issue one K-chunk (32 cols): per thread 2x b128 for A tile + 2x for W tile
  auto issue = [&](int k0, int buf) {
    async_ld_b128(lA[buf],      gA + k0);
    async_ld_b128(lA[buf] + 16, gA + k0 + 8);
    async_ld_b128(lW[buf],      gW + k0);
    async_ld_b128(lW[buf] + 16, gW + k0 + 8);
  };

  v8f acc[2][4] = {};
  const int nk = K >> 5;

  issue(0, 0);  // prologue: chunk 0 -> buffer 0

  for (int i = 0; i < nk; ++i) {
    wait_async0();       // this wave's chunk-i DMA landed in LDS
    __syncthreads();     // ... and everyone else's too
    if (i + 1 < nk) {
      issue((i + 1) << 5, (i + 1) & 1);   // overlap next DMA with this compute
      // pull the K-chunk after next toward L2 (global_prefetch_b8)
      __builtin_prefetch((const void*)(gW + ((i + 2) << 5)), 0, 1);
    }
    const int buf = i & 1;

    union F { v16bf v; unsigned u[8]; uint4 q[2]; };

    // A fragments (16x32 bf16: lane m=L%16, K contiguous 16 per half-wave)
    F afrag[2];
#pragma unroll
    for (int mi = 0; mi < 2; ++mi) {
      const int m = wm + mi * 16 + l16;
#pragma unroll
      for (int v = 0; v < 8; ++v) {
        const int kk = g * 16 + (v << 1);
        afrag[mi].u[v] = *(const unsigned*)&As[buf][m][kk];
      }
    }

#pragma unroll
    for (int ni = 0; ni < 4; ++ni) {
      F bfrag;
      const int n = wn + ni * 16 + l16;
      const uint4* pw = (const uint4*)&Ws[buf][n][g << 4];
      bfrag.q[0] = pw[0];
      bfrag.q[1] = pw[1];
#pragma unroll
      for (int mi = 0; mi < 2; ++mi) {
        acc[mi][ni] = __builtin_amdgcn_wmma_f32_16x16x32_bf16(
            false, afrag[mi].v, false, bfrag.v,
            (short)0, acc[mi][ni], false, false);
      }
    }
  }

  // ---- epilogue: C += acc (C/D layout: VGPR r -> m = r + 8*(lane/16), n = lane%16)
#pragma unroll
  for (int mi = 0; mi < 2; ++mi)
#pragma unroll
    for (int ni = 0; ni < 4; ++ni)
#pragma unroll
      for (int r = 0; r < 8; ++r) {
        const int mm = bm + wm + mi * 16 + r + (g << 3);
        const int nn = bn + wn + ni * 16 + l16;
        C[(size_t)mm * ldc + nn] += acc[mi][ni][r];
      }
}

// ---- fused LSTM cell: gates -> (h,c), plus bf16 h copies for next GEMMs ----

__global__ __launch_bounds__(256)
void lstm_cell(const float* __restrict__ gates,
               float* __restrict__ h, float* __restrict__ c,
               __bf16* __restrict__ dst1, int ld1, int off1,
               __bf16* __restrict__ dst2, int ld2, int off2,
               int Bn, int Hn) {
  int i = blockIdx.x * blockDim.x + threadIdx.x;
  if (i >= Bn * Hn) return;
  int b = i / Hn, j = i % Hn;
  const float* g4 = gates + (size_t)b * 4 * Hn;
  float ig = sigm(g4[j]);
  float fg = sigm(g4[Hn + j]);
  float gg = tanh_fast(g4[2 * Hn + j]);
  float og = sigm(g4[3 * Hn + j]);
  float cn = fg * c[i] + ig * gg;
  float hn = og * tanh_fast(cn);
  c[i] = cn;
  h[i] = hn;
  __bf16 hb = f2bf(hn);
  dst1[(size_t)b * ld1 + off1 + j] = hb;
  if (dst2) dst2[(size_t)b * ld2 + off2 + j] = hb;
}

// ---- greedy argmax over V per batch row -----------------------------------

__global__ __launch_bounds__(256)
void argmax_rows(const float* __restrict__ pred, int ldp, int off,
                 int Vn, int* __restrict__ idx) {
  __shared__ float sv[256];
  __shared__ int   si[256];
  const int b = blockIdx.x;
  const int tid = threadIdx.x;
  float best = -3.4e38f; int bi = 0;
  for (int v = tid; v < Vn; v += 256) {
    float x = pred[(size_t)b * ldp + off + v];
    if (x > best) { best = x; bi = v; }
  }
  sv[tid] = best; si[tid] = bi;
  __syncthreads();
  for (int s = 128; s > 0; s >>= 1) {
    if (tid < s) {
      if (sv[tid + s] > sv[tid] ||
          (sv[tid + s] == sv[tid] && si[tid + s] < si[tid])) {
        sv[tid] = sv[tid + s]; si[tid] = si[tid + s];
      }
    }
    __syncthreads();
  }
  if (tid == 0) idx[b] = si[0];
}

// ---------------------------------------------------------------------------

extern "C" void kernel_launch(void* const* d_in, const int* in_sizes, int n_in,
                              void* d_out, int out_size, void* d_ws, size_t ws_size,
                              hipStream_t stream) {
  const int Bn = 256, Tin = 256, Dn = 512, Hn = 1024, Vn = 512, Tout = 64;
  const int Gn = 4 * Hn;  // 4096

  const float* x     = (const float*)d_in[0];
  const float* eWih0 = (const float*)d_in[1];
  const float* eWhh0 = (const float*)d_in[2];
  const float* eb0   = (const float*)d_in[3];
  const float* eWih1 = (const float*)d_in[4];
  const float* eWhh1 = (const float*)d_in[5];
  const float* eb1   = (const float*)d_in[6];
  const float* dWih0 = (const float*)d_in[7];
  const float* dWhh0 = (const float*)d_in[8];
  const float* db0   = (const float*)d_in[9];
  const float* dWih1 = (const float*)d_in[10];
  const float* dWhh1 = (const float*)d_in[11];
  const float* db1   = (const float*)d_in[12];
  const float* Wout  = (const float*)d_in[13];
  const float* bout  = (const float*)d_in[14];
  float* out = (float*)d_out;

  // ---- workspace carve-out (all 256B aligned) ----
  size_t cur = 0;
  auto alloc = [&](size_t bytes) -> void* {
    cur = (cur + 255) & ~(size_t)255;
    void* r = (char*)d_ws + cur;
    cur += bytes;
    return r;
  };
  // packed bf16 weights (Wih|Whh concatenated along K where fused); ~57MB,
  // resident in the 192MB L2 so every recurrent step streams them from L2.
  __bf16* WcatE0 = (__bf16*)alloc((size_t)Gn * (Dn + Hn) * 2);  // 4096x1536
  __bf16* WcatE1 = (__bf16*)alloc((size_t)Gn * (2 * Hn) * 2);   // 4096x2048
  __bf16* WdIh0  = (__bf16*)alloc((size_t)Gn * Vn * 2);         // gather source
  __bf16* WdHh0  = (__bf16*)alloc((size_t)Gn * Hn * 2);         // 4096x1024
  __bf16* WcatD1 = (__bf16*)alloc((size_t)Gn * (2 * Hn) * 2);   // 4096x2048
  __bf16* WoutB  = (__bf16*)alloc((size_t)Vn * Hn * 2);         // 512x1024
  // bf16 activation (A-matrix) buffers
  __bf16* AeL0 = (__bf16*)alloc((size_t)Bn * (Dn + Hn) * 2);    // [x_t | h0]
  __bf16* AeL1 = (__bf16*)alloc((size_t)Bn * (2 * Hn) * 2);     // [h0  | h1]
  __bf16* AdL0 = (__bf16*)alloc((size_t)Bn * Hn * 2);           // h0 (dec)
  __bf16* AdL1 = (__bf16*)alloc((size_t)Bn * (2 * Hn) * 2);     // [h0 | h1]
  __bf16* Aout = (__bf16*)alloc((size_t)Bn * Hn * 2);           // h1 (proj)
  // fp32 state
  float* gates = (float*)alloc((size_t)Bn * Gn * 4);
  float* h0 = (float*)alloc((size_t)Bn * Hn * 4);
  float* c0 = (float*)alloc((size_t)Bn * Hn * 4);
  float* h1 = (float*)alloc((size_t)Bn * Hn * 4);
  float* c1 = (float*)alloc((size_t)Bn * Hn * 4);
  int* idx = (int*)alloc((size_t)Bn * 4);

  auto blk = [](int n) { return dim3((unsigned)((n + 255) / 256)); };
  const dim3 t256(256);

  // ---- phase 0: pack weights fp32 -> bf16 (HBM read once; bf16 stays in L2)
  cvt_f32_bf16<<<blk(Gn * Dn), t256, 0, stream>>>(WcatE0, Dn + Hn, 0,  eWih0, Dn, 0, Gn, Dn);
  cvt_f32_bf16<<<blk(Gn * Hn), t256, 0, stream>>>(WcatE0, Dn + Hn, Dn, eWhh0, Hn, 0, Gn, Hn);
  cvt_f32_bf16<<<blk(Gn * Hn), t256, 0, stream>>>(WcatE1, 2 * Hn, 0,   eWih1, Hn, 0, Gn, Hn);
  cvt_f32_bf16<<<blk(Gn * Hn), t256, 0, stream>>>(WcatE1, 2 * Hn, Hn,  eWhh1, Hn, 0, Gn, Hn);
  cvt_f32_bf16<<<blk(Gn * Vn), t256, 0, stream>>>(WdIh0,  Vn, 0,       dWih0, Vn, 0, Gn, Vn);
  cvt_f32_bf16<<<blk(Gn * Hn), t256, 0, stream>>>(WdHh0,  Hn, 0,       dWhh0, Hn, 0, Gn, Hn);
  cvt_f32_bf16<<<blk(Gn * Hn), t256, 0, stream>>>(WcatD1, 2 * Hn, 0,   dWih1, Hn, 0, Gn, Hn);
  cvt_f32_bf16<<<blk(Gn * Hn), t256, 0, stream>>>(WcatD1, 2 * Hn, Hn,  dWhh1, Hn, 0, Gn, Hn);
  cvt_f32_bf16<<<blk(Vn * Hn), t256, 0, stream>>>(WoutB,  Hn, 0,       Wout,  Hn, 0, Vn, Hn);

  // zero initial state + recurrent halves of A buffers (t=0 reads them)
  fill_f32<<<blk(Bn * Hn), t256, 0, stream>>>(h0, 0.f, Bn * Hn);
  fill_f32<<<blk(Bn * Hn), t256, 0, stream>>>(c0, 0.f, Bn * Hn);
  fill_f32<<<blk(Bn * Hn), t256, 0, stream>>>(h1, 0.f, Bn * Hn);
  fill_f32<<<blk(Bn * Hn), t256, 0, stream>>>(c1, 0.f, Bn * Hn);
  zero_bf16_strided<<<blk(Bn * Hn), t256, 0, stream>>>(AeL0, Dn + Hn, Dn, Bn, Hn);
  zero_bf16_strided<<<blk(Bn * Hn), t256, 0, stream>>>(AeL1, 2 * Hn, Hn, Bn, Hn);

  const dim3 gGates(Gn / 128, Bn / 128);  // 32 x 2 blocks
  const dim3 gProj(Vn / 128, Bn / 128);   // 4 x 2 blocks

  // ---- phase 1: encoder (both layers fused per timestep; no y_seq buffer) ----
  for (int t = 0; t < Tin; ++t) {
    // A = [bf16(x_t) | h0_prev]
    cvt_f32_bf16<<<blk(Bn * Dn), t256, 0, stream>>>(
        AeL0, Dn + Hn, 0, x, Tin * Dn, t * Dn, Bn, Dn);
    // layer 0: gates = b0 + A @ [Wih0|Whh0]^T
    gates_init_bias<<<blk(Bn * Gn), t256, 0, stream>>>(gates, Gn, eb0, Bn, Gn);
    gemm_bf16_wmma<<<gGates, t256, 0, stream>>>(
        AeL0, Dn + Hn, WcatE0, Dn + Hn, gates, Gn, Bn, Gn, Dn + Hn);
    lstm_cell<<<blk(Bn * Hn), t256, 0, stream>>>(
        gates, h0, c0, AeL0, Dn + Hn, Dn, AeL1, 2 * Hn, 0, Bn, Hn);
    // layer 1: gates = b1 + [h0 | h1_prev] @ [Wih1|Whh1]^T
    gates_init_bias<<<blk(Bn * Gn), t256, 0, stream>>>(gates, Gn, eb1, Bn, Gn);
    gemm_bf16_wmma<<<gGates, t256, 0, stream>>>(
        AeL1, 2 * Hn, WcatE1, 2 * Hn, gates, Gn, Bn, Gn, 2 * Hn);
    lstm_cell<<<blk(Bn * Hn), t256, 0, stream>>>(
        gates, h1, c1, AeL1, 2 * Hn, Hn, (__bf16*)nullptr, 0, 0, Bn, Hn);
  }

  // ---- transition: decoder inherits encoder final (h,c); seed bf16 A bufs ----
  cvt_f32_bf16<<<blk(Bn * Hn), t256, 0, stream>>>(AdL0, Hn, 0,      h0, Hn, 0, Bn, Hn);
  cvt_f32_bf16<<<blk(Bn * Hn), t256, 0, stream>>>(AdL1, 2 * Hn, Hn, h1, Hn, 0, Bn, Hn);

  // ---- phase 2: autoregressive decoder ----
  for (int t = 0; t < Tout; ++t) {
    // layer 0: one-hot input => column gather of Wih0 (no GEMM for Wih part)
    gates_init_gather<<<blk(Bn * Gn), t256, 0, stream>>>(
        gates, Gn, db0, WdIh0, Vn, idx, (t > 0) ? 1 : 0, Bn, Gn);
    gemm_bf16_wmma<<<gGates, t256, 0, stream>>>(
        AdL0, Hn, WdHh0, Hn, gates, Gn, Bn, Gn, Hn);
    lstm_cell<<<blk(Bn * Hn), t256, 0, stream>>>(
        gates, h0, c0, AdL0, Hn, 0, AdL1, 2 * Hn, 0, Bn, Hn);
    // layer 1
    gates_init_bias<<<blk(Bn * Gn), t256, 0, stream>>>(gates, Gn, db1, Bn, Gn);
    gemm_bf16_wmma<<<gGates, t256, 0, stream>>>(
        AdL1, 2 * Hn, WcatD1, 2 * Hn, gates, Gn, Bn, Gn, 2 * Hn);
    lstm_cell<<<blk(Bn * Hn), t256, 0, stream>>>(
        gates, h1, c1, AdL1, 2 * Hn, Hn, Aout, Hn, 0, Bn, Hn);
    // output projection straight into d_out[:, t, :]
    gates_init_bias<<<blk(Bn * Vn), t256, 0, stream>>>(
        out + (size_t)t * Vn, Tout * Vn, bout, Bn, Vn);
    gemm_bf16_wmma<<<gProj, t256, 0, stream>>>(
        Aout, Hn, WoutB, Hn, out + (size_t)t * Vn, Tout * Vn, Bn, Vn, Hn);
    // greedy token -> idx feeds next step's gather
    argmax_rows<<<dim3(Bn), t256, 0, stream>>>(out, Tout * Vn, t * Vn, Vn, idx);
  }
  (void)in_sizes; (void)n_in; (void)out_size; (void)ws_size;
}
